// BaseApproximator_6030134083679
// MI455X (gfx1250) — matrix operand order repaired and checked by reading, not compiled, not run on hardware
//
#include <hip/hip_runtime.h>
#include <hip/hip_bf16.h>

// ---------------------------------------------------------------------------
// Performer random-feature map on MI455X (gfx1250, wave32, WMMA).
//   out[n,s] = exp(-0.5*||x_n||^2) * S^-0.5 * exp( x_n . W_s + b_s )
// GEMM done with split-precision bf16 WMMA: x = xhi+xlo, W = whi+wlo,
// acc += xhi*whi + xhi*wlo + xlo*whi  (~f32 accuracy, 3x bf16 matrix work,
// still far below the WMMA ceiling; whole working set is L2-resident).
// ---------------------------------------------------------------------------

typedef __attribute__((ext_vector_type(16))) __bf16 v16bf;
typedef __attribute__((ext_vector_type(8)))  __bf16 v8bf;
typedef __attribute__((ext_vector_type(4)))  __bf16 v4bf;
typedef __attribute__((ext_vector_type(8)))  float  v8f;

union ABFrag { v16bf v; v8bf h[2]; };

// ---------------------------------------------------------------------------
// Kernel 1: per-row split of x into bf16 hi/lo + row-norm -> hscale.
// One 256-thread block per row (D = 1024 -> one float4 per thread).
// ---------------------------------------------------------------------------
__global__ __launch_bounds__(256)
void prep_x_kernel(const float* __restrict__ x,
                   __bf16* __restrict__ xhi, __bf16* __restrict__ xlo,
                   float* __restrict__ hscale, int D, float inv_sqrt_S) {
    const int row = blockIdx.x;
    const float* xr = x + (size_t)row * D;
    __bf16* hr = xhi + (size_t)row * D;
    __bf16* lr = xlo + (size_t)row * D;

    float ss = 0.0f;
    for (int i = threadIdx.x * 4; i < D; i += blockDim.x * 4) {
        float4 v = *(const float4*)(xr + i);
        float vs[4] = {v.x, v.y, v.z, v.w};
        v4bf hv, lv;
#pragma unroll
        for (int j = 0; j < 4; ++j) {
            __bf16 h = (__bf16)vs[j];              // RTNE f32->bf16
            hv[j] = h;
            lv[j] = (__bf16)(vs[j] - (float)h);    // residual
            ss += vs[j] * vs[j];
        }
        *(v4bf*)(hr + i) = hv;
        *(v4bf*)(lr + i) = lv;
    }

    __shared__ float red[256];
    red[threadIdx.x] = ss;
    __syncthreads();
    for (int off = 128; off > 0; off >>= 1) {
        if (threadIdx.x < off) red[threadIdx.x] += red[threadIdx.x + off];
        __syncthreads();
    }
    if (threadIdx.x == 0) {
        // matches reference: h = exp(-0.5*||x||^2) computed in f32, then * S^-1/2
        hscale[row] = expf(-0.5f * red[0]) * inv_sqrt_S;
    }
}

// ---------------------------------------------------------------------------
// Kernel 2: grid-stride bf16 hi/lo split of W.
// ---------------------------------------------------------------------------
__global__ __launch_bounds__(256)
void prep_w_kernel(const float* __restrict__ W,
                   __bf16* __restrict__ whi, __bf16* __restrict__ wlo,
                   long n_elems) {
    long stride = (long)gridDim.x * blockDim.x * 4;
    for (long i = ((long)blockIdx.x * blockDim.x + threadIdx.x) * 4;
         i < n_elems; i += stride) {
        float4 v = *(const float4*)(W + i);
        float vs[4] = {v.x, v.y, v.z, v.w};
        v4bf hv, lv;
#pragma unroll
        for (int j = 0; j < 4; ++j) {
            __bf16 h = (__bf16)vs[j];
            hv[j] = h;
            lv[j] = (__bf16)(vs[j] - (float)h);
        }
        *(v4bf*)(whi + i) = hv;
        *(v4bf*)(wlo + i) = lv;
    }
}

// ---------------------------------------------------------------------------
// Kernel 3: WMMA GEMM + fused epilogue.
// Block: 256 threads = 8 waves (wave32). Output tile 128(N) x 128(S).
// Wave grid 2(M) x 4(S): each wave -> 64 rows x 32 cols = 4x2 accumulators.
//
// Fragment addressing uses contiguous k order, same for A and B (not the ISA table order):
//  A (16x32 bf16): lane L holds row (L&15); lanes<16 hold K[0..15],
//                  lanes>=16 hold K[16..31]          -> two 16B loads/lane.
//  B (32x16 bf16): lane L holds column (L&15); lanes<16 hold K[0..15],
//                  lanes>=16 hold K[16..31]          -> two 16B loads/lane.
//                  (W is row-major [S,D]: B column s == contiguous W row s.)
//  C (16x16 f32):  VGPR r -> row r + 8*(L>=16), col = L&15.
// ---------------------------------------------------------------------------
__global__ __launch_bounds__(256)
void performer_gemm_kernel(const __bf16* __restrict__ xhi,
                           const __bf16* __restrict__ xlo,
                           const __bf16* __restrict__ whi,
                           const __bf16* __restrict__ wlo,
                           const float* __restrict__ b,
                           const float* __restrict__ hscale,
                           float* __restrict__ out,
                           int D, int S) {
    const int lane = threadIdx.x & 31;
    const int wave = threadIdx.x >> 5;
    const int wm   = wave & 1;     // wave row-group: 64 rows each
    const int wsg  = wave >> 1;    // wave col-group: 32 cols each
    const int half = lane >> 4;    // 0 / 1
    const int l16  = lane & 15;

    const int n0 = blockIdx.y * 128 + wm * 64;   // wave's first output row
    const int s0 = blockIdx.x * 128 + wsg * 32;  // wave's first output col

    v8f acc[4][2] = {};  // 4 m-tiles x 2 s-tiles of 16x16 f32

    for (int k0 = 0; k0 < D; k0 += 32) {
        // ---- B fragments (hi & lo), two 16B contiguous loads each ----
        ABFrag bh[2], bl[2];
#pragma unroll
        for (int st = 0; st < 2; ++st) {
            size_t boff = (size_t)(s0 + st * 16 + l16) * D + k0 + 16 * half;
            bh[st].h[0] = *(const v8bf*)(whi + boff);
            bh[st].h[1] = *(const v8bf*)(whi + boff + 8);
            bl[st].h[0] = *(const v8bf*)(wlo + boff);
            bl[st].h[1] = *(const v8bf*)(wlo + boff + 8);
        }
        // ---- A fragments (hi & lo), two 16B contiguous loads each ----
        ABFrag ah[4], al[4];
#pragma unroll
        for (int mt = 0; mt < 4; ++mt) {
            size_t aoff = (size_t)(n0 + mt * 16 + l16) * D + k0 + 16 * half;
            ah[mt].h[0] = *(const v8bf*)(xhi + aoff);
            ah[mt].h[1] = *(const v8bf*)(xhi + aoff + 8);
            al[mt].h[0] = *(const v8bf*)(xlo + aoff);
            al[mt].h[1] = *(const v8bf*)(xlo + aoff + 8);
        }
        // ---- 24 WMMAs: hi*hi + hi*lo + lo*hi ----
#pragma unroll
        for (int mt = 0; mt < 4; ++mt) {
#pragma unroll
            for (int st = 0; st < 2; ++st) {
                acc[mt][st] = __builtin_amdgcn_wmma_f32_16x16x32_bf16(
                    false, ah[mt].v, false, bh[st].v, (short)0, acc[mt][st],
                    false, false);
                acc[mt][st] = __builtin_amdgcn_wmma_f32_16x16x32_bf16(
                    false, ah[mt].v, false, bl[st].v, (short)0, acc[mt][st],
                    false, false);
                acc[mt][st] = __builtin_amdgcn_wmma_f32_16x16x32_bf16(
                    false, al[mt].v, false, bh[st].v, (short)0, acc[mt][st],
                    false, false);
            }
        }
    }

    // ---- fused epilogue: out = hscale[n] * exp(acc + b[s]) ----
#pragma unroll
    for (int st = 0; st < 2; ++st) {
        const int col = s0 + st * 16 + l16;
        const float bc = b[col];
#pragma unroll
        for (int mt = 0; mt < 4; ++mt) {
            const int rowb = n0 + mt * 16 + 8 * half;        // 8 consecutive rows
            v8f hs = *(const v8f*)(hscale + rowb);           // 32B aligned
            float* op = out + (size_t)rowb * S + col;
#pragma unroll
            for (int r = 0; r < 8; ++r) {
                op[(size_t)r * S] = hs[r] * expf(acc[mt][st][r] + bc);
            }
        }
    }
}

// ---------------------------------------------------------------------------
// Host launcher. Inputs: x [N*D] f32, W [S*D] f32, b [S] f32. Output [N*S] f32.
// Workspace: xhi/xlo (N*D bf16 each), whi/wlo (S*D bf16 each), hscale (N f32)
// ~72 MB total for the reference shapes.
// ---------------------------------------------------------------------------
extern "C" void kernel_launch(void* const* d_in, const int* in_sizes, int n_in,
                              void* d_out, int out_size, void* d_ws, size_t ws_size,
                              hipStream_t stream) {
    const float* x = (const float*)d_in[0];
    const float* W = (const float*)d_in[1];
    const float* b = (const float*)d_in[2];
    float* out = (float*)d_out;

    const long xe = in_sizes[0];
    const long we = in_sizes[1];
    const int  S  = in_sizes[2];
    const int  D  = (int)(we / S);
    const int  N  = (int)(xe / D);

    char* ws = (char*)d_ws;
    size_t off = 0;
    auto carve = [&](size_t bytes) -> void* {
        off = (off + 255) & ~(size_t)255;
        void* p = ws + off;
        off += bytes;
        return p;
    };
    __bf16* xhi = (__bf16*)carve((size_t)N * D * sizeof(__bf16));
    __bf16* xlo = (__bf16*)carve((size_t)N * D * sizeof(__bf16));
    __bf16* whi = (__bf16*)carve((size_t)S * D * sizeof(__bf16));
    __bf16* wlo = (__bf16*)carve((size_t)S * D * sizeof(__bf16));
    float*  hscale = (float*)carve((size_t)N * sizeof(float));
    (void)ws_size; (void)n_in; (void)out_size;

    const float inv_sqrt_S = rsqrtf((float)S);

    prep_x_kernel<<<N, 256, 0, stream>>>(x, xhi, xlo, hscale, D, inv_sqrt_S);

    {
        long n_elems = (long)S * D;
        int blocks = (int)((n_elems / 4 + 255) / 256);
        if (blocks > 1024) blocks = 1024;
        prep_w_kernel<<<blocks, 256, 0, stream>>>(W, whi, wlo, n_elems);
    }

    dim3 grid(S / 128, N / 128);
    performer_gemm_kernel<<<grid, 256, 0, stream>>>(xhi, xlo, whi, wlo, b,
                                                    hscale, out, D, S);
}
